// EncoderSimilarity_19481971655014
// MI455X (gfx1250) — matrix operand order repaired and checked by reading, not compiled, not run on hardware
//
#include <hip/hip_runtime.h>

// ---------------------------------------------------------------------------
// Problem constants (from reference): C=48 captions, N=48 images, W=24 words,
// R=36 regions, D=1024, S=256.  CN = 2304, CNW = 55296.
// ---------------------------------------------------------------------------
#define CC   48
#define NN   48
#define WW   24
#define RR   36
#define DD   1024
#define SS   256
#define CN   (CC * NN)
#define CNW  (CN * WW)

typedef __attribute__((ext_vector_type(16))) _Float16 v16h;
typedef __attribute__((ext_vector_type(8)))  float    v8f;

// ---------------------------------------------------------------------------
// WMMA helpers (CDNA5 gfx1250, wave32).
// A fragment (16x32 f16): lane l holds row m=l&15; VGPR v holds K pair
//   k = (l>>4)*16 + v*2 (+0/1).
// B fragment (32x16 f16) loaded from BT (row-major N x K): lane l holds
//   column n=l&15, K block (l>>4)*16 .. +15 contiguous.
// C/D f32 fragment: acc[v] == element (m = v + 8*(l>>4), n = l&15).
// ---------------------------------------------------------------------------
__device__ __forceinline__ v16h frag_a(const _Float16* M, int ld, int row0,
                                       int k0, int lane) {
  v16h f;
  const _Float16* rp = M + (size_t)(row0 + (lane & 15)) * ld + k0;
  int half = lane >> 4;
#pragma unroll
  for (int v = 0; v < 8; ++v) {
    int k = (half << 4) + (v << 1);
    f[2 * v]     = rp[k];
    f[2 * v + 1] = rp[k + 1];
  }
  return f;
}

__device__ __forceinline__ v16h frag_b(const _Float16* BT, int ld, int n0,
                                       int k0, int lane) {
  v16h f;
  const _Float16* rp =
      BT + (size_t)(n0 + (lane & 15)) * ld + k0 + ((lane >> 4) << 4);
#pragma unroll
  for (int j = 0; j < 16; ++j) f[j] = rp[j];
  return f;
}

__device__ __forceinline__ v8f wmma32(v16h a, v16h b, v8f c) {
  return __builtin_amdgcn_wmma_f32_16x16x32_f16(false, a, false, b, (short)0,
                                                c, false, false);
}

__device__ __forceinline__ v8f zero8() {
  v8f z;
#pragma unroll
  for (int v = 0; v < 8; ++v) z[v] = 0.f;
  return z;
}

// ---------------------------------------------------------------------------
// Prep / packing kernels (f32 -> f16, zero-padded tiles, transposed weights)
// ---------------------------------------------------------------------------
__global__ void k_pack_imgA(const float* __restrict__ img,
                            _Float16* __restrict__ dst) {
  // dst: (N, 48, 1024), rows >= 36 zero  (WMMA A for attn matmul)
  size_t i = (size_t)blockIdx.x * blockDim.x + threadIdx.x;
  size_t total = (size_t)NN * 48 * DD;
  if (i >= total) return;
  int d = i % DD;
  int r = (i / DD) % 48;
  int n = i / (48 * DD);
  float v = (r < RR) ? img[((size_t)n * RR + r) * DD + d] : 0.f;
  dst[i] = (_Float16)v;
}

__global__ void k_pack_imgT(const float* __restrict__ img,
                            _Float16* __restrict__ dst) {
  // dst: (N, 1024, 64) = img^T per image, K(=region) padded 36->64 with zeros
  size_t i = (size_t)blockIdx.x * blockDim.x + threadIdx.x;
  size_t total = (size_t)NN * DD * 64;
  if (i >= total) return;
  int r = i % 64;
  int d = (i / 64) % DD;
  int n = i / (64 * DD);
  float v = (r < RR) ? img[((size_t)n * RR + r) * DD + d] : 0.f;
  dst[i] = (_Float16)v;
}

__global__ void k_pack_cap(const float* __restrict__ cap,
                           _Float16* __restrict__ dst) {
  // dst: (C, 32, 1024), word rows >= 24 zero (acts as BT for attn matmul)
  size_t i = (size_t)blockIdx.x * blockDim.x + threadIdx.x;
  size_t total = (size_t)CC * 32 * DD;
  if (i >= total) return;
  int d = i % DD;
  int w = (i / DD) % 32;
  int c = i / (32 * DD);
  float v = (w < WW) ? cap[((size_t)c * WW + w) * DD + d] : 0.f;
  dst[i] = (_Float16)v;
}

__global__ void k_w_to_f16T(const float* __restrict__ Wm,
                            _Float16* __restrict__ WT, int K, int Ncol) {
  // W: (K, Ncol) f32 row-major  ->  WT: (Ncol, K) f16 row-major
  size_t i = (size_t)blockIdx.x * blockDim.x + threadIdx.x;
  if (i >= (size_t)K * Ncol) return;
  int n = i % Ncol;
  int k = i / Ncol;
  WT[(size_t)n * K + k] = (_Float16)Wm[i];
}

// ---------------------------------------------------------------------------
// Attention matmul 1 + fused epilogue.  One block (64 thr = 2 waves) per (c,n).
// attn_raw(48x32) = imgA(48x1024) x (cap*matrix)^T ; then leaky-relu, word
// mask, l2norm over words, softmax*smooth over regions -> attn2 f16 (32x64),
// zero padded (words>=24, regions>=36).  B fragment shared by 3 M-tiles.
// ---------------------------------------------------------------------------
__global__ void k_attn(const _Float16* __restrict__ imgA,
                       const _Float16* __restrict__ capP,
                       const _Float16* __restrict__ mat16,  // null => 1.0
                       const float* __restrict__ smootharr, // null => 9.0
                       const int* __restrict__ lens,
                       _Float16* __restrict__ attn2) {
  int cn = blockIdx.x, c = cn / NN, n = cn % NN;
  int tid = threadIdx.x, wave = tid >> 5, lane = tid & 31;
  __shared__ float att[48][33];
  __shared__ float smax[WW], ssum[WW], ssm[WW];

  const _Float16* Ab = imgA + (size_t)n * 48 * DD;
  const _Float16* Bb = capP + (size_t)c * 32 * DD;
  int wcol = wave * 16 + (lane & 15);   // word column this lane owns (0..31)

  v8f acc[3];
#pragma unroll
  for (int t = 0; t < 3; ++t) acc[t] = zero8();

  for (int k0 = 0; k0 < DD; k0 += 32) {
    // prefetch next A lines (lowering to global_prefetch_b8)
    if (k0 + 64 < DD)
      __builtin_prefetch(Ab + (size_t)(lane & 15) * DD + k0 + 64, 0, 1);
    // B fragment: qm^T column = cap word row (contiguous), times matrix
    v16h b;
    const _Float16* bp = Bb + (size_t)wcol * DD + k0 + ((lane >> 4) << 4);
#pragma unroll
    for (int j = 0; j < 16; ++j) b[j] = bp[j];
    if (mat16 && wcol < WW) {
      const _Float16* mp =
          mat16 + ((size_t)cn * WW + wcol) * DD + k0 + ((lane >> 4) << 4);
#pragma unroll
      for (int j = 0; j < 16; ++j)
        b[j] = (_Float16)((float)b[j] * (float)mp[j]);
    }
#pragma unroll
    for (int mt = 0; mt < 3; ++mt) {
      v16h a = frag_a(Ab, DD, mt * 16, k0, lane);
      acc[mt] = wmma32(a, b, acc[mt]);
    }
  }

  int len = lens[c];
#pragma unroll
  for (int mt = 0; mt < 3; ++mt)
#pragma unroll
    for (int v = 0; v < 8; ++v) {
      int r = mt * 16 + v + ((lane >> 4) << 3);
      float x = acc[mt][v];
      x = (x >= 0.f) ? x : 0.1f * x;     // LeakyReLU(0.1)
      if (wcol >= len) x = 0.f;          // word mask (also kills pad cols)
      att[r][wcol] = x;
    }
  __syncthreads();

  // l2norm over words per region row
  if (tid < 48) {
    float s = 0.f;
    for (int w = 0; w < WW; ++w) s += att[tid][w] * att[tid][w];
    float inv = 1.f / (sqrtf(s) + 1e-8f);
    for (int w = 0; w < WW; ++w) att[tid][w] *= inv;
  }
  __syncthreads();

  // softmax over regions (r<36) with per-word smooth
  if (tid < WW) {
    float sm = smootharr ? smootharr[(size_t)cn * WW + tid] : 9.0f;
    float mx = -1e30f;
    for (int r = 0; r < RR; ++r) mx = fmaxf(mx, att[r][tid] * sm);
    float s = 0.f;
    for (int r = 0; r < RR; ++r) s += __expf(att[r][tid] * sm - mx);
    smax[tid] = mx; ssum[tid] = s; ssm[tid] = sm;
  }
  __syncthreads();

  for (int idx = tid; idx < 32 * 64; idx += 64) {
    int w = idx >> 6, r = idx & 63;
    float v = 0.f;
    if (w < WW && r < RR)
      v = __expf(att[r][w] * ssm[w] - smax[w]) / ssum[w];
    attn2[(size_t)cn * 2048 + idx] = (_Float16)v;
  }
}

// ---------------------------------------------------------------------------
// Attention matmul 2: wctx(24x1024) = attn2(32x64) x img(64x1024).
// One block (256 thr = 8 waves) per (c,n).  Wave w owns M-tile (w&1) and
// 16 N-tiles; the two A fragments are loaded ONCE and reused for all 32 WMMAs.
// ---------------------------------------------------------------------------
__global__ void k_wctx(const _Float16* __restrict__ attn2,
                       const _Float16* __restrict__ imgT,
                       _Float16* __restrict__ wctx16) {
  int cn = blockIdx.x, n = cn % NN;
  int wave = threadIdx.x >> 5, lane = threadIdx.x & 31;
  const _Float16* Ab = attn2 + (size_t)cn * 2048;  // (32,64)
  const _Float16* Bb = imgT + (size_t)n * DD * 64; // BT: (1024,64)
  int mt = wave & 1;
  int ntbase = (wave >> 1) * 16;
  v16h a0 = frag_a(Ab, 64, mt * 16, 0, lane);
  v16h a1 = frag_a(Ab, 64, mt * 16, 32, lane);
#pragma unroll 1
  for (int i = 0; i < 16; ++i) {
    int nt = ntbase + i;
    v8f acc = zero8();
    acc = wmma32(a0, frag_b(Bb, 64, nt * 16, 0, lane), acc);
    acc = wmma32(a1, frag_b(Bb, 64, nt * 16, 32, lane), acc);
    int d = nt * 16 + (lane & 15);
#pragma unroll
    for (int v = 0; v < 8; ++v) {
      int m = mt * 16 + v + ((lane >> 4) << 3);
      if (m < WW)
        wctx16[((size_t)cn * WW + m) * DD + d] = (_Float16)acc[v];
    }
  }
}

// ---------------------------------------------------------------------------
// X = (cap - l2norm(wctx))^2 -> f16, one block per (c,n,w) row of 1024.
// ---------------------------------------------------------------------------
__global__ void k_make_X(const _Float16* __restrict__ wctx16,
                         const float* __restrict__ cap,
                         _Float16* __restrict__ X16) {
  int row = blockIdx.x;                  // cn*24 + w
  int t = threadIdx.x;                   // 256 threads
  int c = row / (NN * WW);
  int w = row % WW;
  __shared__ float red[256];
  const _Float16* wr = wctx16 + (size_t)row * DD;
  float s = 0.f;
  for (int j = t; j < DD; j += 256) { float v = (float)wr[j]; s += v * v; }
  red[t] = s; __syncthreads();
  for (int o = 128; o > 0; o >>= 1) {
    if (t < o) red[t] += red[t + o];
    __syncthreads();
  }
  float inv = 1.f / (sqrtf(red[0]) + 1e-8f);
  const float* cr = cap + ((size_t)c * WW + w) * DD;
  for (int j = t; j < DD; j += 256) {
    float d = cr[j] - (float)wr[j] * inv;
    X16[(size_t)row * DD + j] = (_Float16)(d * d);
  }
}

// ---------------------------------------------------------------------------
// Generic WMMA GEMM: C(M,Ncols) = act(A(M,K) x BT^T + bias).
// grid = (M/16, ceil(Ncols/256)); block = 256 (8 waves).  Each wave computes
// TWO adjacent 16x16 tiles so the A fragment feeds 2 WMMAs, and the next
// K-step's fragments are software double-buffered to overlap loads with math.
// act: 0 none, 1 tanh, 2 tanh then clip(x+1, -1, 1) (matrix update).
// ---------------------------------------------------------------------------
__global__ void k_gemm(const _Float16* __restrict__ A,
                       const _Float16* __restrict__ BT,
                       const float* __restrict__ bias,
                       float* __restrict__ C32, _Float16* __restrict__ C16,
                       int Ncols, int K, int act) {
  int mt = blockIdx.x;
  int wave = threadIdx.x >> 5, lane = threadIdx.x & 31;
  int nt0 = (blockIdx.y * 8 + wave) * 2;      // first of two n-tiles
  if (nt0 * 16 >= Ncols) return;              // wave-uniform guard
  int m0 = mt * 16;
  int n0 = nt0 * 16;

  v8f acc0 = zero8(), acc1 = zero8();
  v16h a  = frag_a(A, K, m0, 0, lane);
  v16h b0 = frag_b(BT, K, n0, 0, lane);
  v16h b1 = frag_b(BT, K, n0 + 16, 0, lane);

  for (int k0 = 0; k0 < K; k0 += 32) {
    int kn = k0 + 32;
    v16h an = a, b0n = b0, b1n = b1;
    if (kn < K) {
      if (kn + 32 < K)   // prefetch next-next A line (global_prefetch_b8)
        __builtin_prefetch(A + (size_t)(m0 + (lane & 15)) * K + kn + 32, 0, 1);
      an  = frag_a(A, K, m0, kn, lane);
      b0n = frag_b(BT, K, n0, kn, lane);
      b1n = frag_b(BT, K, n0 + 16, kn, lane);
    }
    acc0 = wmma32(a, b0, acc0);
    acc1 = wmma32(a, b1, acc1);
    a = an; b0 = b0n; b1 = b1n;
  }

  int nc = lane & 15;
#pragma unroll
  for (int t = 0; t < 2; ++t) {
    v8f acc = t ? acc1 : acc0;
    int n = n0 + t * 16 + nc;
    float bv = bias ? bias[n] : 0.f;
#pragma unroll
    for (int v = 0; v < 8; ++v) {
      int m = m0 + v + ((lane >> 4) << 3);
      float x = acc[v] + bv;
      if (act == 1) x = tanhf(x);
      else if (act == 2) { x = tanhf(x) + 1.f; x = x > 1.f ? 1.f : x; }
      if (C32) C32[(size_t)m * Ncols + n] = x;
      if (C16) C16[(size_t)m * Ncols + n] = (_Float16)x;
    }
  }
}

// ---------------------------------------------------------------------------
// Row l2norm (in place f32, plus f16 copy): one block per row.
// ---------------------------------------------------------------------------
__global__ void k_l2norm_rows(float* __restrict__ X,
                              _Float16* __restrict__ X16, int ncols) {
  int row = blockIdx.x, t = threadIdx.x;
  __shared__ float red[256];
  float* rp = X + (size_t)row * ncols;
  float s = 0.f;
  for (int j = t; j < ncols; j += 256) { float v = rp[j]; s += v * v; }
  red[t] = s; __syncthreads();
  for (int o = 128; o > 0; o >>= 1) {
    if (t < o) red[t] += red[t + o];
    __syncthreads();
  }
  float inv = 1.f / (sqrtf(red[0]) + 1e-8f);
  for (int j = t; j < ncols; j += 256) {
    float v = rp[j] * inv;
    rp[j] = v;
    if (X16) X16[(size_t)row * ncols + j] = (_Float16)v;
  }
}

// sim_hig = masked mean of sim_mid over words; one block per (c,n), S threads
__global__ void k_hig_mean(const float* __restrict__ simmid,
                           const int* __restrict__ lens,
                           float* __restrict__ hig,
                           _Float16* __restrict__ hig16) {
  int cn = blockIdx.x, t = threadIdx.x;
  int len = lens[cn / NN];
  float s = 0.f;
  for (int w = 0; w < len; ++w)
    s += simmid[((size_t)cn * WW + w) * SS + t];
  float v = s / (float)len;
  hig[(size_t)cn * SS + t] = v;
  hig16[(size_t)cn * SS + t] = (_Float16)v;
}

// smooth = relu(SM1 @ sm_w2 + sm_b2 + 9); one thread per (c,n,w)
__global__ void k_smooth(const float* __restrict__ SM1,
                         const float* __restrict__ w2,
                         const float* __restrict__ b2,
                         float* __restrict__ smooth) {
  size_t id = (size_t)blockIdx.x * blockDim.x + threadIdx.x;
  if (id >= (size_t)CNW) return;
  float s = 0.f;
  const float* rp = SM1 + id * 128;
  for (int j = 0; j < 128; ++j) s += rp[j] * w2[j];
  s += b2[0] + 9.0f;
  smooth[id] = s > 0.f ? s : 0.f;
}

// RAR attention over words + l2norm; one block per (c,n), S threads
__global__ void k_rar(const float* __restrict__ MIDK,
                      const float* __restrict__ HIGQ,
                      const float* __restrict__ simmid,
                      const float* __restrict__ vw, const float* __restrict__ vb,
                      const int* __restrict__ lens,
                      float* __restrict__ hig, _Float16* __restrict__ hig16) {
  int cn = blockIdx.x, t = threadIdx.x;
  int len = lens[cn / NN];
  __shared__ float red[256];
  __shared__ float ww[WW];
  float hq = HIGQ[(size_t)cn * SS + t] * vw[t];
  for (int w = 0; w < WW; ++w) {
    red[t] = MIDK[((size_t)cn * WW + w) * SS + t] * hq;
    __syncthreads();
    for (int o = 128; o > 0; o >>= 1) {
      if (t < o) red[t] += red[t + o];
      __syncthreads();
    }
    if (t == 0) {
      float s = red[0] + vb[0];
      ww[w] = (w < len) ? s : -1e30f;
    }
    __syncthreads();
  }
  if (t == 0) {
    float mx = -1e30f;
    for (int w = 0; w < WW; ++w) mx = fmaxf(mx, ww[w]);
    float s = 0.f;
    for (int w = 0; w < WW; ++w) { float e = __expf(ww[w] - mx); ww[w] = e; s += e; }
    float inv = 1.f / s;
    for (int w = 0; w < WW; ++w) ww[w] *= inv;
  }
  __syncthreads();
  float o = 0.f;
  for (int w = 0; w < WW; ++w)
    o += ww[w] * simmid[((size_t)cn * WW + w) * SS + t];
  red[t] = o * o; __syncthreads();
  for (int off = 128; off > 0; off >>= 1) {
    if (t < off) red[t] += red[t + off];
    __syncthreads();
  }
  float inv = 1.f / (sqrtf(red[0]) + 1e-8f);
  float v = o * inv;
  hig[(size_t)cn * SS + t] = v;
  hig16[(size_t)cn * SS + t] = (_Float16)v;
}

// final: out[n,c] = sigmoid(sim_hig[c,n] . sim_eval_w + b)
__global__ void k_final(const float* __restrict__ hig,
                        const float* __restrict__ ew,
                        const float* __restrict__ eb,
                        float* __restrict__ out) {
  int id = blockIdx.x * blockDim.x + threadIdx.x;
  if (id >= CN) return;
  int c = id / NN, n = id % NN;
  float s = eb[0];
  const float* rp = hig + (size_t)id * SS;
  for (int j = 0; j < SS; ++j) s += rp[j] * ew[j];
  out[(size_t)n * CC + c] = 1.f / (1.f + expf(-s));
}

// ---------------------------------------------------------------------------
// Host launcher
// ---------------------------------------------------------------------------
extern "C" void kernel_launch(void* const* d_in, const int* in_sizes, int n_in,
                              void* d_out, int out_size, void* d_ws,
                              size_t ws_size, hipStream_t stream) {
  const float* img    = (const float*)d_in[0];   // (N,R,D)
  const float* cap    = (const float*)d_in[1];   // (C,W,D)
  const int*   lens   = (const int*)d_in[2];     // (C,)
  const float* alv_w  = (const float*)d_in[3];   // (2,D,S)
  const float* alv_b  = (const float*)d_in[4];   // (2,S)
  const float* q_w    = (const float*)d_in[5];   // (2,S,S)
  const float* q_b    = (const float*)d_in[6];
  const float* kk_w   = (const float*)d_in[7];   // (2,S,S)
  const float* kk_b   = (const float*)d_in[8];
  const float* v_w    = (const float*)d_in[9];   // (2,S,1)
  const float* v_b    = (const float*)d_in[10];  // (2,1)
  const float* sm_w1  = (const float*)d_in[11];  // (S,128)
  const float* sm_b1  = (const float*)d_in[12];
  const float* sm_w2  = (const float*)d_in[13];  // (128,1)
  const float* sm_b2  = (const float*)d_in[14];
  const float* mx_w1  = (const float*)d_in[15];  // (S,512)
  const float* mx_b1  = (const float*)d_in[16];
  const float* mx_w2  = (const float*)d_in[17];  // (512,D)
  const float* mx_b2  = (const float*)d_in[18];
  const float* ev_w   = (const float*)d_in[19];  // (S,1)
  const float* ev_b   = (const float*)d_in[20];
  float* out = (float*)d_out;

  // bump allocator over workspace
  char* base = (char*)d_ws;
  size_t off = 0;
  auto alloc = [&](size_t bytes) -> void* {
    off = (off + 255) & ~(size_t)255;
    void* p = base + off;
    off += bytes;
    return p;
  };
  _Float16* imgA   = (_Float16*)alloc((size_t)NN * 48 * DD * 2);
  _Float16* imgT   = (_Float16*)alloc((size_t)NN * DD * 64 * 2);
  _Float16* capP   = (_Float16*)alloc((size_t)CC * 32 * DD * 2);
  _Float16* alvT   = (_Float16*)alloc((size_t)2 * SS * DD * 2);
  _Float16* qT     = (_Float16*)alloc((size_t)2 * SS * SS * 2);
  _Float16* kT     = (_Float16*)alloc((size_t)2 * SS * SS * 2);
  _Float16* mx1T   = (_Float16*)alloc((size_t)512 * SS * 2);
  _Float16* mx2T   = (_Float16*)alloc((size_t)DD * 512 * 2);
  _Float16* sm1T   = (_Float16*)alloc((size_t)128 * SS * 2);
  _Float16* attn2  = (_Float16*)alloc((size_t)CN * 2048 * 2);
  _Float16* wctx16 = (_Float16*)alloc((size_t)CNW * DD * 2);
  _Float16* X16    = (_Float16*)alloc((size_t)CNW * DD * 2);
  float*    P      = (float*)  alloc((size_t)CNW * SS * 4);   // sim_mid f32
  _Float16* P16    = (_Float16*)alloc((size_t)CNW * SS * 2);
  _Float16* H116   = (_Float16*)alloc((size_t)CNW * 512 * 2);
  _Float16* MAT16  = (_Float16*)alloc((size_t)CNW * DD * 2);  // matrix
  float*    SM1    = (float*)  alloc((size_t)CNW * 128 * 4);
  float*    SMOOTH = (float*)  alloc((size_t)CNW * 4);
  float*    MIDK   = (float*)  alloc((size_t)CNW * SS * 4);
  float*    HIGQ   = (float*)  alloc((size_t)CN * SS * 4);
  float*    HIG    = (float*)  alloc((size_t)CN * SS * 4);
  _Float16* HIG16  = (_Float16*)alloc((size_t)CN * SS * 2);
  (void)ws_size; (void)in_sizes; (void)n_in; (void)out_size;

  auto g1 = [](size_t n) { return dim3((unsigned)((n + 255) / 256)); };
  auto gy = [](int ncols) { return (unsigned)((ncols + 255) / 256); };

  // ---- prep: pack f16 operands, transpose weights
  k_pack_imgA<<<g1((size_t)NN * 48 * DD), 256, 0, stream>>>(img, imgA);
  k_pack_imgT<<<g1((size_t)NN * DD * 64), 256, 0, stream>>>(img, imgT);
  k_pack_cap <<<g1((size_t)CC * 32 * DD), 256, 0, stream>>>(cap, capP);
  for (int m = 0; m < 2; ++m) {
    k_w_to_f16T<<<g1((size_t)DD * SS), 256, 0, stream>>>(
        alv_w + (size_t)m * DD * SS, alvT + (size_t)m * SS * DD, DD, SS);
    k_w_to_f16T<<<g1((size_t)SS * SS), 256, 0, stream>>>(
        q_w + (size_t)m * SS * SS, qT + (size_t)m * SS * SS, SS, SS);
    k_w_to_f16T<<<g1((size_t)SS * SS), 256, 0, stream>>>(
        kk_w + (size_t)m * SS * SS, kT + (size_t)m * SS * SS, SS, SS);
  }
  k_w_to_f16T<<<g1((size_t)SS * 512), 256, 0, stream>>>(mx_w1, mx1T, SS, 512);
  k_w_to_f16T<<<g1((size_t)512 * DD), 256, 0, stream>>>(mx_w2, mx2T, 512, DD);
  k_w_to_f16T<<<g1((size_t)SS * 128), 256, 0, stream>>>(sm_w1, sm1T, SS, 128);

  const int MT = CNW / 16;  // 3456 M-tiles for the big batched GEMMs

  for (int m = 0; m < 2; ++m) {
    // ---- xattn + alv(m)
    k_attn<<<CN, 64, 0, stream>>>(imgA, capP, m ? MAT16 : nullptr,
                                  m ? SMOOTH : nullptr, lens, attn2);
    k_wctx<<<CN, 256, 0, stream>>>(attn2, imgT, wctx16);
    k_make_X<<<CNW, 256, 0, stream>>>(wctx16, cap, X16);
    k_gemm<<<dim3(MT, gy(SS)), 256, 0, stream>>>(
        X16, alvT + (size_t)m * SS * DD, alv_b + (size_t)m * SS, P, nullptr,
        SS, DD, 0);
    k_l2norm_rows<<<CNW, 256, 0, stream>>>(P, P16, SS);  // -> sim_mid

    if (m == 0) {
      // ---- sim_hig = masked mean; RCR matrix + smooth
      k_hig_mean<<<CN, SS, 0, stream>>>(P, lens, HIG, HIG16);
      k_gemm<<<dim3(MT, gy(512)), 256, 0, stream>>>(
          P16, mx1T, mx_b1, nullptr, H116, 512, SS, 1);          // tanh
      k_gemm<<<dim3(MT, gy(DD)), 256, 0, stream>>>(
          H116, mx2T, mx_b2, nullptr, MAT16, DD, 512, 2);        // tanh+clip+1
      k_gemm<<<dim3(MT, gy(128)), 256, 0, stream>>>(
          P16, sm1T, sm_b1, SM1, nullptr, 128, SS, 1);           // tanh
      k_smooth<<<g1(CNW), 256, 0, stream>>>(SM1, sm_w2, sm_b2, SMOOTH);
    }

    // ---- rar(m)
    k_gemm<<<dim3(MT, gy(SS)), 256, 0, stream>>>(
        P16, kT + (size_t)m * SS * SS, kk_b + (size_t)m * SS, MIDK, nullptr,
        SS, SS, 1);                                              // mid_k
    k_gemm<<<dim3(CN / 16, gy(SS)), 256, 0, stream>>>(
        HIG16, qT + (size_t)m * SS * SS, q_b + (size_t)m * SS, HIGQ, nullptr,
        SS, SS, 1);                                              // hig_q
    k_rar<<<CN, SS, 0, stream>>>(MIDK, HIGQ, P, v_w + (size_t)m * SS,
                                 v_b + m, lens, HIG, HIG16);
  }

  // ---- final sigmoid + transpose to (N, C)
  k_final<<<g1(CN), 256, 0, stream>>>(HIG, ev_w, ev_b, out);
}
